// SelfAttention_12824772346405
// MI455X (gfx1250) — matrix operand order repaired and checked by reading, not compiled, not run on hardware
//
#include <hip/hip_runtime.h>
#include <hip/hip_bf16.h>

#define SEQ    2048
#define DIN    1024
#define DPROJ  128
#define NBATCH 8

typedef __bf16 bf16_t;
typedef __attribute__((ext_vector_type(16))) __bf16 v16bf;
typedef __attribute__((ext_vector_type(8)))  __bf16 v8bf;
typedef __attribute__((ext_vector_type(8)))  float  v8f;
typedef __attribute__((ext_vector_type(4)))  float  f32x4;
typedef __attribute__((ext_vector_type(4)))  int    v4i;

// ---------------- CDNA5 async global->LDS path (guarded) --------------------
#if defined(__has_builtin)
#if __has_builtin(__builtin_amdgcn_global_load_async_to_lds_b128)
#define HAVE_ASYNC_LDS 1
#endif
#endif

#ifdef HAVE_ASYNC_LDS
typedef __attribute__((address_space(1))) v4i as1_v4i;
typedef __attribute__((address_space(3))) v4i as3_v4i;
static __device__ __forceinline__ void async16(const void* g, void* l) {
  // 16 bytes, memory -> LDS, tracked by ASYNCcnt
  __builtin_amdgcn_global_load_async_to_lds_b128((as1_v4i*)g, (as3_v4i*)l, 0, 0);
}
#if __has_builtin(__builtin_amdgcn_s_wait_asynccnt)
#define ASYNC_WAIT() __builtin_amdgcn_s_wait_asynccnt(0)
#else
#define ASYNC_WAIT() asm volatile("s_wait_asynccnt 0" ::: "memory")
#endif
#else
static __device__ __forceinline__ void async16(const void* g, void* l) {
  *(v8bf*)l = *(const v8bf*)g;   // fallback: load + ds_store
}
#define ASYNC_WAIT() do {} while (0)
#endif

static __device__ __forceinline__ void stage32(const void* g, void* l) {
  async16(g, l);
  async16((const char*)g + 16, (char*)l + 16);
}

// ---------------------------------------------------------------------------
static __device__ __forceinline__ v16bf bf16cat(v8bf lo, v8bf hi) {
  v16bf r;
#pragma unroll
  for (int j = 0; j < 8; ++j) { r[j] = lo[j]; r[8 + j] = hi[j]; }
  return r;
}

static __device__ __forceinline__ v16bf cvt16(f32x4 a, f32x4 b, f32x4 c, f32x4 d) {
  v16bf r;
#pragma unroll
  for (int j = 0; j < 4; ++j) {
    r[j]      = (__bf16)a[j];
    r[4 + j]  = (__bf16)b[j];
    r[8 + j]  = (__bf16)c[j];
    r[12 + j] = (__bf16)d[j];
  }
  return r;
}

static __device__ __forceinline__ v16bf ldsfrag(const bf16_t* p) {
  v8bf lo = *(const v8bf*)p;
  v8bf hi = *(const v8bf*)(p + 8);
  return bf16cat(lo, hi);
}

#define WMMA_BF16(A, B, C) \
  __builtin_amdgcn_wmma_f32_16x16x32_bf16(false, (A), false, (B), (short)0, (C), false, false)

// ---------------------------------------------------------------------------
// Phase 0: one-time f32 -> bf16 weight conversion (Wb[p][col][d], 768 KB).
// ---------------------------------------------------------------------------
__global__ __launch_bounds__(256) void wconv_kernel(
    const float* __restrict__ Wq, const float* __restrict__ Wk,
    const float* __restrict__ Wv, bf16_t* __restrict__ Wb)
{
  const size_t per  = (size_t)DPROJ * DIN;                       // 131072
  const size_t base = ((size_t)blockIdx.x * 256 + threadIdx.x) * 8;
  const float* src = (base < per) ? Wq : (base < 2 * per ? Wk : Wv);
  const size_t off = base & (per - 1);
  f32x4 a = *(const f32x4*)(src + off);
  f32x4 b = *(const f32x4*)(src + off + 4);
  v8bf o;
#pragma unroll
  for (int j = 0; j < 4; ++j) { o[j] = (__bf16)a[j]; o[4 + j] = (__bf16)b[j]; }
  *(v8bf*)(Wb + base) = o;
}

// ---------------------------------------------------------------------------
// Phase 1: fused Q/K/V projection. 8 waves/block share one projection's
// weights; 32-d weight chunks double-buffered in LDS via async-to-LDS.
// Q gets 1/sqrt(128) folded in. V stored transposed: Vt[n][vcol][key].
// ---------------------------------------------------------------------------
#define WS_STRIDE 40   // 32 + 8 bf16 pad (80 B rows, 16B-multiple)

__global__ __launch_bounds__(256) void proj_kernel(
    const float* __restrict__ q_in, const float* __restrict__ k_in,
    const float* __restrict__ v_in, const bf16_t* __restrict__ Wb,
    const float* __restrict__ bq, const float* __restrict__ bk,
    const float* __restrict__ bv,
    bf16_t* __restrict__ Qb, bf16_t* __restrict__ Kb, bf16_t* __restrict__ Vt)
{
  __shared__ alignas(16) bf16_t Ws[2][DPROJ * WS_STRIDE];  // 2 x 10 KB

  const int tid  = threadIdx.x;
  const int lane = tid & 31;
  const int widx = tid >> 5;
  const int w    = blockIdx.x * 8 + widx;   // 0..3071
  const int p    = w >> 10;                 // 0=q,1=k,2=v (uniform per block)
  const int rem  = w & 1023;
  const int n    = rem >> 7;
  const int rt   = rem & 127;

  const float* X; const float* B;
  if (p == 0)      { X = q_in; B = bq; }
  else if (p == 1) { X = k_in; B = bk; }
  else             { X = v_in; B = bv; }
  const bf16_t* Wp = Wb + (size_t)p * DPROJ * DIN;

  const int half = lane >> 4;
  const int l16  = lane & 15;
  const float* xrow = X + ((size_t)n * SEQ + rt * 16 + l16) * DIN;

  // cooperative weight-chunk stage: 128 cols x 32 d (8 KB), 32 B per thread
  const int wcol = tid >> 1, wseg = tid & 1;
  const bf16_t* wsrc = Wp + (size_t)wcol * DIN + wseg * 16;

  stage32(wsrc, &Ws[0][wcol * WS_STRIDE + wseg * 16]);

  v8f acc[8] = {};
  for (int kk = 0; kk < DIN; kk += 32) {
    const int buf = (kk >> 5) & 1;

    // issue A-tile global loads BEFORE the barrier so they fly during it
    const int d0 = kk + half * 16;
    f32x4 a0 = *(const f32x4*)(xrow + d0);
    f32x4 a1 = *(const f32x4*)(xrow + d0 + 4);
    f32x4 a2 = *(const f32x4*)(xrow + d0 + 8);
    f32x4 a3 = *(const f32x4*)(xrow + d0 + 12);

    ASYNC_WAIT();
    __syncthreads();
    if (kk + 32 < DIN)
      stage32(wsrc + kk + 32, &Ws[buf ^ 1][wcol * WS_STRIDE + wseg * 16]);

    v16bf a = cvt16(a0, a1, a2, a3);

    // preload ALL 8 B-fragments, then back-to-back WMMAs (one dscnt wait)
    v16bf bfrag[8];
#pragma unroll
    for (int c = 0; c < 8; ++c)
      bfrag[c] = ldsfrag(&Ws[buf][(c * 16 + l16) * WS_STRIDE + half * 16]);
#pragma unroll
    for (int c = 0; c < 8; ++c)
      acc[c] = WMMA_BF16(a, bfrag[c], acc[c]);
  }

  const float qscale = 0.08838834764831845f;  // 1/sqrt(128)
#pragma unroll
  for (int c = 0; c < 8; ++c) {
    const int col  = c * 16 + l16;
    const float bias = B[col];
    if (p == 2) {
      bf16_t* orow = Vt + ((size_t)n * DPROJ + col) * SEQ + rt * 16 + half * 8;
#pragma unroll
      for (int r = 0; r < 8; ++r)
        orow[r] = (bf16_t)(acc[c][r] + bias);
    } else {
      bf16_t* obase = (p == 0 ? Qb : Kb) +
                      ((size_t)n * SEQ + rt * 16 + half * 8) * DPROJ + col;
      const float s = (p == 0) ? qscale : 1.0f;
#pragma unroll
      for (int r = 0; r < 8; ++r)
        obase[(size_t)r * DPROJ] = (bf16_t)((acc[c][r] + bias) * s);
    }
  }
}

// ---------------------------------------------------------------------------
// Phase 2: flash-attention. 8 waves/block share (batch, key-chunk) K/V tiles
// staged in LDS (async, double-buffered); each wave owns one 16-query tile.
// ---------------------------------------------------------------------------
#define KS_STRIDE 136  // 128 + 8 bf16 pad (272 B rows, 16B-multiple)
#define VS_STRIDE 40   // 32 + 8 bf16 pad

__global__ __launch_bounds__(256) void attn_kernel(
    const unsigned char* __restrict__ mask,
    const bf16_t* __restrict__ Qb, const bf16_t* __restrict__ Kb,
    const bf16_t* __restrict__ Vt, float* __restrict__ out)
{
  __shared__ alignas(16) bf16_t Ks[2][32 * KS_STRIDE];     // 2 x 8.5 KB
  __shared__ alignas(16) bf16_t Vs[2][DPROJ * VS_STRIDE];  // 2 x 10 KB
  __shared__ alignas(16) float  pbuf[8][16][36];           // 18 KB

  const int tid  = threadIdx.x;
  const int lane = tid & 31;
  const int widx = tid >> 5;
  const int n    = blockIdx.x >> 4;                 // batch (uniform per block)
  const int qt   = (blockIdx.x & 15) * 8 + widx;    // 16-query tile
  const int half = lane >> 4;
  const int l16  = lane & 15;

  const bf16_t* Kb_n = Kb + (size_t)n * SEQ * DPROJ;
  const bf16_t* Vt_n = Vt + (size_t)n * DPROJ * SEQ;

  // cooperative stage assignments (32 B per thread per tensor)
  const int krow = tid >> 3, kseg = tid & 7;        // K: 32 rows x 128 d
  const int vcol = tid >> 1, vseg = tid & 1;        // V: 128 cols x 32 keys

  // Q A-fragments (scale folded in), kept resident
  v16bf qa[4];
  const bf16_t* qrow = Qb + ((size_t)n * SEQ + qt * 16 + l16) * DPROJ;
#pragma unroll
  for (int f = 0; f < 4; ++f)
    qa[f] = bf16cat(*(const v8bf*)(qrow + f * 32 + half * 16),
                    *(const v8bf*)(qrow + f * 32 + half * 16 + 8));

  v8f O[8] = {};
  float mrow[8], lrow[8];
#pragma unroll
  for (int r = 0; r < 8; ++r) { mrow[r] = -3.0e38f; lrow[r] = 0.0f; }

  const unsigned char* mbase = mask + ((size_t)n * SEQ + qt * 16) * SEQ;

  // prologue: stage chunk 0
  stage32(Kb_n + (size_t)krow * DPROJ + kseg * 16,
          &Ks[0][krow * KS_STRIDE + kseg * 16]);
  stage32(Vt_n + (size_t)vcol * SEQ + vseg * 16,
          &Vs[0][vcol * VS_STRIDE + vseg * 16]);

  for (int kc = 0; kc < SEQ / 32; ++kc) {
    const int kbase = kc * 32;
    const int buf   = kc & 1;

    ASYNC_WAIT();          // this wave's stage of chunk kc is in LDS
    __syncthreads();       // all waves' stages done; prev buffer fully read

    if (kc + 1 < SEQ / 32) {
      const int nb = kbase + 32;
      stage32(Kb_n + (size_t)(nb + krow) * DPROJ + kseg * 16,
              &Ks[buf ^ 1][krow * KS_STRIDE + kseg * 16]);
      stage32(Vt_n + (size_t)vcol * SEQ + nb + vseg * 16,
              &Vs[buf ^ 1][vcol * VS_STRIDE + vseg * 16]);
    }

    // hoist mask byte loads so they overlap the score WMMAs
    unsigned char mb0[8], mb1[8];
#pragma unroll
    for (int r = 0; r < 8; ++r) {
      const unsigned char* mp = mbase + (size_t)(r + 8 * half) * SEQ + kbase + l16;
      mb0[r] = mp[0];
      mb1[r] = mp[16];
    }

    // ---- S = Q K^T for 32 keys: preload all 8 K-fragments, then 8 WMMAs ----
    const bf16_t* krow0 = &Ks[buf][l16 * KS_STRIDE + half * 16];
    const bf16_t* krow1 = krow0 + 16 * KS_STRIDE;
    v16bf kf[8];
#pragma unroll
    for (int f = 0; f < 4; ++f) {
      kf[f]     = ldsfrag(krow0 + f * 32);
      kf[4 + f] = ldsfrag(krow1 + f * 32);
    }
    v8f S0 = {}, S1 = {};
#pragma unroll
    for (int f = 0; f < 4; ++f) {
      S0 = WMMA_BF16(qa[f], kf[f], S0);
      S1 = WMMA_BF16(qa[f], kf[4 + f], S1);
    }

    // ---- mask + online softmax (row = r + 8*half) ----
#pragma unroll
    for (int r = 0; r < 8; ++r) {
      const int qm = r + 8 * half;
      float s0 = (mb0[r] != 0) ? -1.0e9f : S0[r];
      float s1 = (mb1[r] != 0) ? -1.0e9f : S1[r];
      float cm = fmaxf(s0, s1);
      cm = fmaxf(cm, __shfl_xor(cm, 1, 32));
      cm = fmaxf(cm, __shfl_xor(cm, 2, 32));
      cm = fmaxf(cm, __shfl_xor(cm, 4, 32));
      cm = fmaxf(cm, __shfl_xor(cm, 8, 32));
      const float mnew  = fmaxf(mrow[r], cm);
      const float alpha = __expf(mrow[r] - mnew);
      const float p0 = __expf(s0 - mnew);
      const float p1 = __expf(s1 - mnew);
      float rs = p0 + p1;
      rs += __shfl_xor(rs, 1, 32);
      rs += __shfl_xor(rs, 2, 32);
      rs += __shfl_xor(rs, 4, 32);
      rs += __shfl_xor(rs, 8, 32);
      lrow[r] = lrow[r] * alpha + rs;
      mrow[r] = mnew;
#pragma unroll
      for (int ct = 0; ct < 8; ++ct) O[ct][r] *= alpha;
      pbuf[widx][qm][l16]      = p0;   // C-layout -> LDS (wave-private)
      pbuf[widx][qm][16 + l16] = p1;
    }

    // ---- re-read P as 16x32 bf16 A-fragment ----
    const float* prow = &pbuf[widx][l16][0];
    v16bf pa = cvt16(*(const f32x4*)(prow + half * 16),
                     *(const f32x4*)(prow + half * 16 + 4),
                     *(const f32x4*)(prow + half * 16 + 8),
                     *(const f32x4*)(prow + half * 16 + 12));

    // ---- O += P V : preload all 8 V-fragments, then 8 WMMAs ----
    v16bf vf[8];
#pragma unroll
    for (int ct = 0; ct < 8; ++ct)
      vf[ct] = ldsfrag(&Vs[buf][(ct * 16 + l16) * VS_STRIDE + half * 16]);
#pragma unroll
    for (int ct = 0; ct < 8; ++ct)
      O[ct] = WMMA_BF16(pa, vf[ct], O[ct]);
  }

  // ---- normalize and write out (f32) ----
  float* obase = out + ((size_t)n * SEQ + qt * 16) * DPROJ;
#pragma unroll
  for (int r = 0; r < 8; ++r) {
    const int qm  = r + 8 * half;
    const float inv = 1.0f / lrow[r];
#pragma unroll
    for (int ct = 0; ct < 8; ++ct)
      obase[(size_t)qm * DPROJ + ct * 16 + l16] = O[ct][r] * inv;
  }
}

// ---------------------------------------------------------------------------
extern "C" void kernel_launch(void* const* d_in, const int* in_sizes, int n_in,
                              void* d_out, int out_size, void* d_ws, size_t ws_size,
                              hipStream_t stream) {
  const float* q_in = (const float*)d_in[0];
  const float* k_in = (const float*)d_in[1];
  const float* v_in = (const float*)d_in[2];
  const unsigned char* mask = (const unsigned char*)d_in[3];  // bool array
  const float* Wq = (const float*)d_in[4];
  const float* bq = (const float*)d_in[5];
  const float* Wk = (const float*)d_in[6];
  const float* bk = (const float*)d_in[7];
  const float* Wv = (const float*)d_in[8];
  const float* bv = (const float*)d_in[9];

  const size_t elems = (size_t)NBATCH * SEQ * DPROJ;  // 2M elems per tensor
  bf16_t* Qb = (bf16_t*)d_ws;
  bf16_t* Kb = Qb + elems;
  bf16_t* Vt = Kb + elems;
  bf16_t* Wb = Vt + elems;   // 3 x 128 x 1024 bf16 (768 KB); total ~13.3 MB

  wconv_kernel<<<dim3(192), dim3(256), 0, stream>>>(Wq, Wk, Wv, Wb);

  proj_kernel<<<dim3(384), dim3(256), 0, stream>>>(
      q_in, k_in, v_in, Wb, bq, bk, bv, Qb, Kb, Vt);

  attn_kernel<<<dim3(128), dim3(256), 0, stream>>>(mask, Qb, Kb, Vt, (float*)d_out);

  (void)in_sizes; (void)n_in; (void)out_size; (void)ws_size;
}
